// NWNet_66331474920045
// MI455X (gfx1250) — compile-verified, matrix operand order checked by reading, not run on hardware
//
#include <hip/hip_runtime.h>
#include <hip/hip_bf16.h>
#include <math.h>

// ---------------------------------------------------------------------------
// Problem constants (from reference)
// ---------------------------------------------------------------------------
#define BQ   128
#define SREF 8192
#define MTOT 8320            // BQ + SREF
#define FIN  4096
#define FD   2048
#define PD   1024
#define NC   1000
#define EPSL 1e-12f

typedef __bf16 bf16_t;
typedef __attribute__((ext_vector_type(16))) __bf16 v16bf;
typedef __attribute__((ext_vector_type(8)))  float  v8f;

#if __has_builtin(__builtin_amdgcn_tensor_load_to_lds)
#define HAVE_TDM 1
typedef __attribute__((ext_vector_type(4))) unsigned int u32x4;
typedef __attribute__((ext_vector_type(8))) int          i32x8;
typedef __attribute__((ext_vector_type(4))) int          i32x4;
#endif

// ---------------------------------------------------------------------------
// 1) concat(x, sx) -> bf16 (row-major M x K)
// ---------------------------------------------------------------------------
__global__ __launch_bounds__(256)
void k_concat_bf16(const float* __restrict__ x, const float* __restrict__ sx,
                   bf16_t* __restrict__ out, long n_x, long n_total) {
    long stride = (long)gridDim.x * blockDim.x * 4;
    for (long i = ((long)blockIdx.x * blockDim.x + threadIdx.x) * 4; i < n_total; i += stride) {
        const float* src = (i < n_x) ? (x + i) : (sx + (i - n_x));
        float4 v = *(const float4*)src;
        out[i + 0] = (bf16_t)v.x;
        out[i + 1] = (bf16_t)v.y;
        out[i + 2] = (bf16_t)v.z;
        out[i + 3] = (bf16_t)v.w;
    }
}

// ---------------------------------------------------------------------------
// 2) f32 (rows x cols) -> bf16 transposed (cols x rows), 32x32 LDS tiles
// ---------------------------------------------------------------------------
__global__ __launch_bounds__(256)
void k_transpose_bf16(const float* __restrict__ in, bf16_t* __restrict__ out,
                      int rows, int cols) {
    __shared__ float t[32][33];
    int bx = blockIdx.x * 32;   // col tile
    int by = blockIdx.y * 32;   // row tile
    int tx = threadIdx.x;       // 0..31
    int ty = threadIdx.y;       // 0..7
    #pragma unroll
    for (int i = 0; i < 32; i += 8)
        t[ty + i][tx] = in[(size_t)(by + ty + i) * cols + (bx + tx)];
    __syncthreads();
    #pragma unroll
    for (int i = 0; i < 32; i += 8)
        out[(size_t)(bx + ty + i) * rows + (by + tx)] = (bf16_t)t[tx][ty + i];
}

// ---------------------------------------------------------------------------
// 3) NT GEMM via v_wmma_f32_16x16x32_bf16, tiles staged with the Tensor Data
//    Mover (tensor_load_to_lds, TENSORcnt) and double-buffered in LDS.
//    C(MxN) = A(MxK) * Bt(NxK)^T, M,N multiples of 128, K multiple of 32.
//    256 threads = 8 waves; block tile 128x128; wave tile 64(M) x 32(N).
// ---------------------------------------------------------------------------
#define LDSTRIDE 48   // bf16 elems per LDS row: 64B tile data + 32B pad = 96B

__device__ __forceinline__ unsigned lds_byte_off(const void* p) {
    // generic pointer to a __shared__ object: low 32 bits = LDS byte address
    return (unsigned)(unsigned long long)(uintptr_t)p;
}

#ifdef HAVE_TDM
// One 2D TDM descriptor: tile = 32 elems (64B) x 128 rows of bf16, row stride
// K elems in global memory; LDS padded +8 dwords every 16 dwords (96B rows).
__device__ __forceinline__ void tdm_load_tile(const bf16_t* gsrc, unsigned lds_addr, int K) {
    unsigned long long ga = (unsigned long long)(uintptr_t)gsrc;
    u32x4 g0;
    g0[0] = 1u;                                             // count=1, user mode
    g0[1] = lds_addr;                                       // LDS byte address
    g0[2] = (unsigned)ga;                                   // global_addr[31:0]
    g0[3] = (unsigned)((ga >> 32) & 0x01FFFFFFu) | (2u << 30); // addr[56:32] | type=2

    const unsigned td0 = (unsigned)K;                       // tensor_dim0 (elems)
    const unsigned td1 = 1u << 16;                          // tensor_dim1 (rows, generous)
    const unsigned long long st0 = (unsigned long long)K;   // tensor_dim0_stride (elems)
    // flags: data_size=1 (2B) <<16 | pad_enable<<20 | pad_interval=3 (16 dw) <<22
    //        | pad_amount=7 (8 dw) <<25
    const unsigned flags = (1u << 16) | (1u << 20) | (3u << 22) | (7u << 25);
    i32x8 g1;
    g1[0] = (int)flags;
    g1[1] = (int)((td0 & 0xFFFFu) << 16);                   // atomic_barrier=0 | td0 lo
    g1[2] = (int)((td0 >> 16) | ((td1 & 0xFFFFu) << 16));   // td0 hi | td1 lo
    g1[3] = (int)((td1 >> 16) | (32u << 16));               // td1 hi | tile_dim0=32
    g1[4] = (int)128;                                       // tile_dim1=128, tile_dim2=0
    g1[5] = (int)(unsigned)(st0 & 0xFFFFFFFFull);           // stride0 lo
    g1[6] = (int)(unsigned)((st0 >> 32) & 0xFFFFull);       // stride0 hi | stride1 lo(=0)
    g1[7] = 0;                                              // stride1 hi
    i32x4 z4 = {0, 0, 0, 0};
#if defined(__clang_major__) && (__clang_major__ >= 23)
    i32x8 z8 = {0, 0, 0, 0, 0, 0, 0, 0};
    __builtin_amdgcn_tensor_load_to_lds(g0, g1, z4, z4, z8, 0);
#else
    __builtin_amdgcn_tensor_load_to_lds(g0, g1, z4, z4, 0);
#endif
}

__device__ __forceinline__ void tdm_wait0() {
#if __has_builtin(__builtin_amdgcn_s_wait_tensorcnt)
    __builtin_amdgcn_s_wait_tensorcnt((short)0);
#else
    asm volatile("s_wait_tensorcnt 0x0" ::: "memory");
#endif
}
#endif // HAVE_TDM

template <bool OUT_BF16>
__global__ __launch_bounds__(256)
void k_gemm_nt_wmma(const bf16_t* __restrict__ A, const bf16_t* __restrict__ Bt,
                    void* __restrict__ Cout, int M, int N, int K) {
    __shared__ __attribute__((aligned(64))) bf16_t Al[2][128 * LDSTRIDE];
    __shared__ __attribute__((aligned(64))) bf16_t Bl[2][128 * LDSTRIDE];

    const int tid  = threadIdx.x;
    const int lane = tid & 31;
    const int wave = tid >> 5;
    const int m0   = blockIdx.x * 128;
    const int n0   = blockIdx.y * 128;
    const int mw   = (wave & 1) * 64;    // wave M offset in block
    const int nw   = (wave >> 1) * 32;   // wave N offset in block
    const int r16  = lane & 15;
    const int half = lane >> 4;

    v8f acc[4][2];
    #pragma unroll
    for (int i = 0; i < 4; ++i)
        #pragma unroll
        for (int j = 0; j < 2; ++j) {
            v8f z = {0.f, 0.f, 0.f, 0.f, 0.f, 0.f, 0.f, 0.f};
            acc[i][j] = z;
        }

    const int nk = K >> 5;               // K-steps of 32

#ifdef HAVE_TDM
    // ---------------- async TDM double-buffered pipeline -----------------
    if (wave == 0) {
        tdm_load_tile(A  + (size_t)m0 * K, lds_byte_off(&Al[0][0]), K);
        tdm_load_tile(Bt + (size_t)n0 * K, lds_byte_off(&Bl[0][0]), K);
    }
    for (int kt = 0; kt < nk; ++kt) {
        const int cur = kt & 1;
        if (wave == 0) tdm_wait0();      // buf[cur] DMA complete
        __syncthreads();                 // visible to all waves; prev compute done
        if (wave == 0 && (kt + 1) < nk) {
            tdm_load_tile(A  + (size_t)m0 * K + (size_t)(kt + 1) * 32,
                          lds_byte_off(&Al[cur ^ 1][0]), K);
            tdm_load_tile(Bt + (size_t)n0 * K + (size_t)(kt + 1) * 32,
                          lds_byte_off(&Bl[cur ^ 1][0]), K);
        }
        // ---- fragments (ISA 16-bit A layout: two 8-elem K runs per lane) ----
        v16bf afr[4], bfr[2];
        #pragma unroll
        for (int i = 0; i < 4; ++i) {
            const bf16_t* p = &Al[cur][(mw + i * 16 + r16) * LDSTRIDE + half * 8];
            union { v16bf v; uint4 q[2]; } u;
            u.q[0] = *(const uint4*)p;
            u.q[1] = *(const uint4*)(p + 16);
            afr[i] = u.v;
        }
        #pragma unroll
        for (int j = 0; j < 2; ++j) {
            const bf16_t* p = &Bl[cur][(nw + j * 16 + r16) * LDSTRIDE + half * 8];
            union { v16bf v; uint4 q[2]; } u;
            u.q[0] = *(const uint4*)p;
            u.q[1] = *(const uint4*)(p + 16);
            bfr[j] = u.v;
        }
        #pragma unroll
        for (int i = 0; i < 4; ++i)
            #pragma unroll
            for (int j = 0; j < 2; ++j)
                acc[i][j] = __builtin_amdgcn_wmma_f32_16x16x32_bf16(
                    false, afr[i], false, bfr[j], (short)0, acc[i][j], false, false);
    }
#else
    // ---------------- fallback: cooperative VMEM -> LDS ------------------
    for (int kt = 0; kt < nk; ++kt) {
        const int k0 = kt * 32;
        #pragma unroll
        for (int t = 0; t < 2; ++t) {
            int idx = tid + t * 256;
            int row = idx >> 2;
            int kq  = (idx & 3) * 8;
            *(uint4*)&Al[0][row * LDSTRIDE + kq] =
                *(const uint4*)(A  + (size_t)(m0 + row) * K + k0 + kq);
            *(uint4*)&Bl[0][row * LDSTRIDE + kq] =
                *(const uint4*)(Bt + (size_t)(n0 + row) * K + k0 + kq);
        }
        if (k0 + 32 < K) {
            __builtin_prefetch(A  + (size_t)(m0 + (tid >> 1)) * K + k0 + 32, 0, 1);
            __builtin_prefetch(Bt + (size_t)(n0 + (tid >> 1)) * K + k0 + 32, 0, 1);
        }
        __syncthreads();
        v16bf afr[4], bfr[2];
        #pragma unroll
        for (int i = 0; i < 4; ++i) {
            const bf16_t* p = &Al[0][(mw + i * 16 + r16) * LDSTRIDE + half * 8];
            union { v16bf v; uint4 q[2]; } u;
            u.q[0] = *(const uint4*)p;
            u.q[1] = *(const uint4*)(p + 16);
            afr[i] = u.v;
        }
        #pragma unroll
        for (int j = 0; j < 2; ++j) {
            const bf16_t* p = &Bl[0][(nw + j * 16 + r16) * LDSTRIDE + half * 8];
            union { v16bf v; uint4 q[2]; } u;
            u.q[0] = *(const uint4*)p;
            u.q[1] = *(const uint4*)(p + 16);
            bfr[j] = u.v;
        }
        #pragma unroll
        for (int i = 0; i < 4; ++i)
            #pragma unroll
            for (int j = 0; j < 2; ++j)
                acc[i][j] = __builtin_amdgcn_wmma_f32_16x16x32_bf16(
                    false, afr[i], false, bfr[j], (short)0, acc[i][j], false, false);
        __syncthreads();
    }
#endif

    // ---- epilogue: C layout = lane N = lane&15, M = r + (lane>>4)*8 ----
    #pragma unroll
    for (int i = 0; i < 4; ++i)
        #pragma unroll
        for (int j = 0; j < 2; ++j) {
            int col = n0 + nw + j * 16 + r16;
            #pragma unroll
            for (int r = 0; r < 8; ++r) {
                int row = m0 + mw + i * 16 + half * 8 + r;
                float v = acc[i][j][r];
                if (OUT_BF16)
                    ((bf16_t*)Cout)[(size_t)row * N + col] = (bf16_t)v;
                else
                    ((float*)Cout)[(size_t)row * N + col] = v;
            }
        }
}

// ---------------------------------------------------------------------------
// 4) row squared-norms of bf16 matrix (one block of 256 per row)
// ---------------------------------------------------------------------------
__global__ __launch_bounds__(256)
void k_row_norms(const bf16_t* __restrict__ m, float* __restrict__ norms, int K) {
    __shared__ float red[8];
    int row = blockIdx.x;
    float s = 0.f;
    for (int k = threadIdx.x; k < K; k += 256) {
        float v = (float)m[(size_t)row * K + k];
        s += v * v;
    }
    #pragma unroll
    for (int d = 16; d > 0; d >>= 1) s += __shfl_xor(s, d, 32);
    int lane = threadIdx.x & 31, w = threadIdx.x >> 5;
    if (lane == 0) red[w] = s;
    __syncthreads();
    if (threadIdx.x == 0) {
        float tot = 0.f;
        #pragma unroll
        for (int i = 0; i < 8; ++i) tot += red[i];
        norms[row] = tot;
    }
}

// ---------------------------------------------------------------------------
// 5) fused distance -> softmax -> class scatter -> log  (1 block per query)
// ---------------------------------------------------------------------------
__global__ __launch_bounds__(1024)
void k_softmax_scatter(const float* __restrict__ dots,     // BQ x SREF  (q @ s^T)
                       const float* __restrict__ norms,    // MTOT (q first, then s)
                       const int*   __restrict__ sy,       // SREF
                       float* __restrict__ out) {          // BQ x NC
    __shared__ float cls[NC];
    __shared__ float red[32];

    int b   = blockIdx.x;
    int tid = threadIdx.x;
    const float  qn    = norms[b];
    const float* snorm = norms + BQ;
    const float* drow  = dots + (size_t)b * SREF;

    for (int c = tid; c < NC; c += 1024) cls[c] = 0.f;

    // pass 1: max score
    float mx = -__builtin_inff();
    for (int n = tid; n < SREF; n += 1024) {
        float sq = qn + snorm[n] - 2.f * drow[n];
        float sc = -sqrtf(fmaxf(sq, 0.f));
        mx = fmaxf(mx, sc);
    }
    #pragma unroll
    for (int d = 16; d > 0; d >>= 1) mx = fmaxf(mx, __shfl_xor(mx, d, 32));
    int lane = tid & 31, w = tid >> 5;
    if (lane == 0) red[w] = mx;
    __syncthreads();
    if (w == 0) {
        float r = red[lane];
        #pragma unroll
        for (int d = 16; d > 0; d >>= 1) r = fmaxf(r, __shfl_xor(r, d, 32));
        if (lane == 0) red[0] = r;
    }
    __syncthreads();
    const float bm = red[0];
    __syncthreads();

    // pass 2: sum of exp
    float z = 0.f;
    for (int n = tid; n < SREF; n += 1024) {
        float sq = qn + snorm[n] - 2.f * drow[n];
        float sc = -sqrtf(fmaxf(sq, 0.f));
        z += __expf(sc - bm);
    }
    #pragma unroll
    for (int d = 16; d > 0; d >>= 1) z += __shfl_xor(z, d, 32);
    if (lane == 0) red[w] = z;
    __syncthreads();
    if (w == 0) {
        float r = red[lane];
        #pragma unroll
        for (int d = 16; d > 0; d >>= 1) r += __shfl_xor(r, d, 32);
        if (lane == 0) red[0] = r;
    }
    __syncthreads();
    const float Z = red[0];

    // pass 3: scatter exp into class bins (LDS ds_add_f32 atomics)
    for (int n = tid; n < SREF; n += 1024) {
        float sq = qn + snorm[n] - 2.f * drow[n];
        float sc = -sqrtf(fmaxf(sq, 0.f));
        atomicAdd(&cls[sy[n]], __expf(sc - bm));
    }
    __syncthreads();

    const float invZ = 1.f / Z;
    for (int c = tid; c < NC; c += 1024)
        out[(size_t)b * NC + c] = logf(cls[c] * invZ + EPSL);
}

// ---------------------------------------------------------------------------
// launch
// ---------------------------------------------------------------------------
extern "C" void kernel_launch(void* const* d_in, const int* in_sizes, int n_in,
                              void* d_out, int out_size, void* d_ws, size_t ws_size,
                              hipStream_t stream) {
    const float* x    = (const float*)d_in[0];   // 128 x 4096
    const float* sx   = (const float*)d_in[1];   // 8192 x 4096
    const int*   sy   = (const int*)  d_in[2];   // 8192
    const float* Wf   = (const float*)d_in[3];   // 4096 x 2048
    const float* Pw   = (const float*)d_in[4];   // 2048 x 1024
    float*       out  = (float*)d_out;           // 128 x 1000

    // workspace layout (all 16B-aligned offsets)
    char* w = (char*)d_ws;
    constexpr size_t OFF_XCAT  = 0;                                   // 8320*4096*2
    constexpr size_t OFF_WFT   = OFF_XCAT + (size_t)MTOT * FIN * 2;   // 2048*4096*2
    constexpr size_t OFF_PRT   = OFF_WFT  + (size_t)FD  * FIN * 2;    // 1024*2048*2
    constexpr size_t OFF_FEATS = OFF_PRT  + (size_t)PD  * FD  * 2;    // 8320*2048*2
    constexpr size_t OFF_QS    = OFF_FEATS+ (size_t)MTOT * FD * 2;    // 8320*1024*2
    constexpr size_t OFF_NORM  = OFF_QS   + (size_t)MTOT * PD * 2;    // 8320*4
    constexpr size_t OFF_DOTS  = OFF_NORM + (size_t)MTOT * 4;         // 128*8192*4

    bf16_t* xcat  = (bf16_t*)(w + OFF_XCAT);
    bf16_t* WfT   = (bf16_t*)(w + OFF_WFT);
    bf16_t* PwT   = (bf16_t*)(w + OFF_PRT);
    bf16_t* feats = (bf16_t*)(w + OFF_FEATS);
    bf16_t* qs    = (bf16_t*)(w + OFF_QS);
    float*  norms = (float*) (w + OFF_NORM);
    float*  dots  = (float*) (w + OFF_DOTS);

    // 1) concat + convert to bf16
    k_concat_bf16<<<4096, 256, 0, stream>>>(x, sx, xcat,
                                            (long)BQ * FIN, (long)MTOT * FIN);

    // 2) transposed bf16 weights
    k_transpose_bf16<<<dim3(FD / 32, FIN / 32), dim3(32, 8), 0, stream>>>(Wf, WfT, FIN, FD);
    k_transpose_bf16<<<dim3(PD / 32, FD  / 32), dim3(32, 8), 0, stream>>>(Pw, PwT, FD, PD);

    // 3) GEMM1: feats(8320x2048) = xcat(8320x4096) * WfT(2048x4096)^T
    k_gemm_nt_wmma<true><<<dim3(MTOT / 128, FD / 128), 256, 0, stream>>>(
        xcat, WfT, feats, MTOT, FD, FIN);

    // 4) GEMM2: qs(8320x1024) = feats * PwT^T
    k_gemm_nt_wmma<true><<<dim3(MTOT / 128, PD / 128), 256, 0, stream>>>(
        feats, PwT, qs, MTOT, PD, FD);

    // 5) row norms of q|s
    k_row_norms<<<MTOT, 256, 0, stream>>>(qs, norms, PD);

    // 6) GEMM3: dots(128x8192) = q(128x1024) * s(8192x1024)^T  (f32 out)
    k_gemm_nt_wmma<false><<<dim3(BQ / 128, SREF / 128), 256, 0, stream>>>(
        qs, qs + (size_t)BQ * PD, dots, BQ, SREF, PD);

    // 7) fused softmax + class scatter + log
    k_softmax_scatter<<<BQ, 1024, 0, stream>>>(dots, norms, sy, out);
}
